// ESN_Cell_8048768712769
// MI455X (gfx1250) — matrix operand order repaired and checked by reading, not compiled, not run on hardware
//
#include <hip/hip_runtime.h>

typedef __attribute__((ext_vector_type(16))) __bf16 v16bf;
typedef __attribute__((ext_vector_type(8)))  float  v8f;
typedef __attribute__((ext_vector_type(4)))  int    v4i;

#if defined(__has_builtin)
#  if __has_builtin(__builtin_amdgcn_global_load_async_to_lds_b128)
#    define USE_ASYNC_LDS 1
#  endif
#  if __has_builtin(__builtin_amdgcn_s_wait_asynccnt)
#    define HAVE_WAIT_ASYNC_BUILTIN 1
#  endif
#endif

typedef __attribute__((address_space(1))) v4i* gptr_v4i;   // global (HIP "__device__")
typedef __attribute__((address_space(3))) v4i* lptr_v4i;   // LDS    (HIP "__shared__")

union Frag {
  v16bf v;
  uint4 q[2];
};

__device__ __forceinline__ unsigned short f2bf(float f) {
  unsigned int u = __float_as_uint(f);
  unsigned int r = u + 0x7FFFu + ((u >> 16) & 1u);   // round-to-nearest-even
  return (unsigned short)(r >> 16);
}

__device__ __forceinline__ void wait_async0() {
#if defined(HAVE_WAIT_ASYNC_BUILTIN)
  __builtin_amdgcn_s_wait_asynccnt(0);
#else
  asm volatile("s_wait_asynccnt 0" ::: "memory");
#endif
}

// ---------------------------------------------------------------------------
// Pass 1: pack Win/Wres (f32, HWIO) into bf16 K-blocks in d_ws.
// Layout: wB[kb][cout][k], kb = 0..63, cout = 0..63, k = 0..31.
//   kb <  50 : reservoir path. tap = kb/2, c = (kb&1)*32 + k, Wres[tap][c][cout]
//   kb >= 50 : input path. tap pair (2*(kb-50), +1); k<16 -> tap0, else tap1;
//              cslot = k&15; cslot<4 -> Win[tap][cslot][cout], else 0.
//   kb == 63 : all zero (K-loop padding).
// ---------------------------------------------------------------------------
__global__ __launch_bounds__(256) void esn_pack_weights(
    const float* __restrict__ Win, const float* __restrict__ Wres,
    unsigned short* __restrict__ wB) {
  int idx = blockIdx.x * 256 + threadIdx.x;
  if (idx >= 64 * 2048) return;
  int kb   = idx >> 11;
  int rem  = idx & 2047;
  int cout = rem >> 5;
  int k    = rem & 31;
  float val = 0.0f;
  if (kb < 50) {
    int tap = kb >> 1;
    int c   = ((kb & 1) << 5) + k;
    val = Wres[(tap * 64 + c) * 64 + cout];
  } else {
    int b2  = kb - 50;
    int tap = 2 * b2 + (k >> 4);
    int cs  = k & 15;
    if (tap < 25 && cs < 4) val = Win[(tap * 4 + cs) * 64 + cout];
  }
  wB[idx] = f2bf(val);
}

// ---------------------------------------------------------------------------
// Pass 2: one workgroup per (b, h): M=128 pixels x N=64 couts implicit GEMM.
// act slot stride padded 80 -> 88 bf16 (176 B = 44 banks, gcd(44,64)=4)
// to reduce LDS bank-conflict degree on A-fragment loads.
// ---------------------------------------------------------------------------
#define ACT_STRIDE 88
#define WBUF_OFF   116160          // act: 5*132*88 bf16 = 116160 B
#define ZPAD_OFF   148928          // wbuf: 8 blocks * 4096 B = 32768 B
#define SMEM_SZ    148960          // + 32 B zero pad

__global__ __launch_bounds__(256) void esn_main(
    const float* __restrict__ inputs, const float* __restrict__ states,
    const unsigned short* __restrict__ wB, float* __restrict__ out) {
  __shared__ __align__(16) unsigned char smem[SMEM_SZ];
  unsigned short* act  = (unsigned short*)smem;
  unsigned short* wbuf = (unsigned short*)(smem + WBUF_OFF);
  unsigned short* zpad = (unsigned short*)(smem + ZPAD_OFF);
  float* outs = (float*)smem;      // aliases act after the K loop is done

  const int tid = threadIdx.x;
  const int b = blockIdx.x >> 7;
  const int h = blockIdx.x & 127;

  if (tid < 2) ((uint4*)zpad)[tid] = make_uint4(0u, 0u, 0u, 0u);

  // --- stage activations: [kh][wi][ACT_STRIDE] bf16, channel-contiguous.
  //     wi 0..131 covers circularly padded columns; 8 channels per iteration
  //     packed into one 16B ds_store (global reads stay coalesced along w).
  for (int i = tid; i < 5 * 10 * 132; i += 256) {
    int kh = i / 1320;
    int r  = i - kh * 1320;
    int cg = r / 132;
    int wi = r - cg * 132;
    int row = (h + kh + 126) & 127;
    int col = (wi + 126) & 127;
    unsigned int pk[4];
    if (cg < 8) {                               // 64 reservoir channels
      const float* sp = states + ((b * 64 + cg * 8) * 128 + row) * 128 + col;
#pragma unroll
      for (int j = 0; j < 4; ++j) {
        unsigned short lo = f2bf(sp[(2 * j) * 16384]);
        unsigned short up = f2bf(sp[(2 * j + 1) * 16384]);
        pk[j] = (unsigned int)lo | ((unsigned int)up << 16);
      }
    } else if (cg == 8) {                       // 4 input channels + 4 zeros
      const float* ip = inputs + ((b * 4) * 128 + row) * 128 + col;
      unsigned short v0 = f2bf(ip[0]);
      unsigned short v1 = f2bf(ip[16384]);
      unsigned short v2 = f2bf(ip[2 * 16384]);
      unsigned short v3 = f2bf(ip[3 * 16384]);
      pk[0] = (unsigned int)v0 | ((unsigned int)v1 << 16);
      pk[1] = (unsigned int)v2 | ((unsigned int)v3 << 16);
      pk[2] = 0u; pk[3] = 0u;
    } else {                                    // 8 zero-pad channels
      pk[0] = pk[1] = pk[2] = pk[3] = 0u;
    }
    *(uint4*)(act + (kh * 132 + wi) * ACT_STRIDE + cg * 8) =
        make_uint4(pk[0], pk[1], pk[2], pk[3]);
  }
  __syncthreads();

  const int lane = tid & 31;
  const int wv   = tid >> 5;
  const int l15  = lane & 15;
  const bool hih = lane >= 16;
  const int w0   = wv * 16 + l15;    // A-matrix row (pixel)
  const int oA   = hih ? 8 : 0;      // A lane K-chunk offsets: {oA, oA+16}
  const int oB   = hih ? 8 : 0;      // B lane K-chunk offsets: {oB, oB+16}

  v8f acc[4] = {};

  for (int cc = 0; cc < 8; ++cc) {
    // stage 8 weight blocks (32 KB) into LDS, cooperatively
#if defined(USE_ASYNC_LDS)
    {
      // CDNA5 async DMA: global -> LDS, tracked by ASYNCcnt (no VGPR bounce).
      uint4* src = (uint4*)(wB + cc * 8 * 2048);
      uint4* dst = (uint4*)wbuf;
#pragma unroll
      for (int i = 0; i < 8; ++i) {
        __builtin_amdgcn_global_load_async_to_lds_b128(
            (gptr_v4i)(src + tid + i * 256),
            (lptr_v4i)(dst + tid + i * 256),
            0, 0);
      }
      wait_async0();
    }
#else
    {
      const uint4* src = (const uint4*)(wB + cc * 8 * 2048);
      uint4* dst = (uint4*)wbuf;
#pragma unroll
      for (int i = 0; i < 8; ++i) dst[tid + i * 256] = src[tid + i * 256];
    }
#endif
    __syncthreads();

#pragma unroll
    for (int kk = 0; kk < 8; ++kk) {
      const int kb = cc * 8 + kk;
      Frag A;
      if (kb < 50) {                            // reservoir path
        int tap = kb >> 1;
        int kh  = tap / 5;
        int kw  = tap - kh * 5;
        int cb  = (kb & 1) << 5;
        const unsigned short* p = act + (kh * 132 + w0 + kw) * ACT_STRIDE + cb + oA;
        A.q[0] = *(const uint4*)p;
        A.q[1] = *(const uint4*)(p + 16);
      } else {                                  // input path (tap pairs)
        int t0 = 2 * (kb - 50);
        int t1 = t0 + 1;
        if (t0 < 25) {
          int kh = t0 / 5, kw = t0 - kh * 5;
          A.q[0] = *(const uint4*)(act + (kh * 132 + w0 + kw) * ACT_STRIDE + 64 + oA);
        } else {
          A.q[0] = *(const uint4*)zpad;
        }
        if (t1 < 25) {
          int kh = t1 / 5, kw = t1 - kh * 5;
          A.q[1] = *(const uint4*)(act + (kh * 132 + w0 + kw) * ACT_STRIDE + 64 + oA);
        } else {
          A.q[1] = *(const uint4*)zpad;
        }
      }
      const unsigned short* wp = wbuf + kk * 2048 + l15 * 32 + oB;
#pragma unroll
      for (int nt = 0; nt < 4; ++nt) {
        Frag Bf;
        Bf.q[0] = *(const uint4*)(wp + nt * 512);
        Bf.q[1] = *(const uint4*)(wp + nt * 512 + 16);
        acc[nt] = __builtin_amdgcn_wmma_f32_16x16x32_bf16(
            false, A.v, false, Bf.v, (short)0, acc[nt], false, false);
      }
    }
    __syncthreads();
  }

  // --- epilogue: transpose accumulators through LDS for coalesced I/O.
  // C/D layout: VGPR r, lanes 0-15: M=r, N=lane; lanes 16-31: M=r+8, N=lane-16.
#pragma unroll
  for (int nt = 0; nt < 4; ++nt) {
    int cout = nt * 16 + l15;
#pragma unroll
    for (int r = 0; r < 8; ++r) {
      int w = wv * 16 + (hih ? r + 8 : r);
      outs[cout * 128 + w] = acc[nt][r];
    }
  }
  __syncthreads();

  const int base = ((b * 64) * 128 + h) * 128;
  for (int i = tid; i < 64 * 128; i += 256) {
    int cout = i >> 7;
    int w    = i & 127;
    int g    = base + cout * 16384 + w;
    float pre  = outs[i];
    float s    = states[g];
    float cand = tanhf(pre);
    out[g] = s + 0.8f * (cand - s);     // leaky integration, ALPHA = 0.8
  }
}

// ---------------------------------------------------------------------------
extern "C" void kernel_launch(void* const* d_in, const int* in_sizes, int n_in,
                              void* d_out, int out_size, void* d_ws, size_t ws_size,
                              hipStream_t stream) {
  const float* inputs = (const float*)d_in[0];   // [16, 4, 128, 128]
  const float* states = (const float*)d_in[1];   // [16, 64, 128, 128]
  const float* Win    = (const float*)d_in[2];   // [5, 5, 4, 64]
  const float* Wres   = (const float*)d_in[3];   // [5, 5, 64, 64]
  float* out = (float*)d_out;
  unsigned short* wB = (unsigned short*)d_ws;    // 64*2048 bf16 = 256 KB

  esn_pack_weights<<<512, 256, 0, stream>>>(Win, Wres, wB);
  esn_main<<<2048, 256, 0, stream>>>(inputs, states, wB, out);
}
